// QueryPlanPredictor_53300544143942
// MI455X (gfx1250) — compile-verified, matrix operand order checked by reading, not run on hardware
//
#include <hip/hip_runtime.h>
#include <hip/hip_bf16.h>
#include <stdint.h>

typedef _Float16 f16;
typedef _Float16 v16h __attribute__((ext_vector_type(16)));
typedef float    v8f  __attribute__((ext_vector_type(8)));

#define LAYERS   2
#define DMODEL   2048
#define NHEAD    16
#define HEADD    128
#define FFN      5504
#define BATCH    4
#define SEQ      1024
#define LRANK    8
#define MTOK     (BATCH * SEQ)
#define LORA_SCALE 4.0f                  // ALPHA / R = 32 / 8
#define RMS_EPS    1e-5f
#define INV_SQRT_HD 0.08838834764831845f // 1/sqrt(128)
#define LOG_THETA   9.210340371976184f   // ln(10000)

// --- CDNA5 async global->LDS copy support (compile-safe detection) ---------
#if defined(__gfx1250__) && __has_builtin(__builtin_amdgcn_global_load_async_to_lds_b128)
#define HAVE_ASYNC_LDS 1
typedef int v4i __attribute__((vector_size(16)));          // b128 payload type
typedef __attribute__((address_space(3))) v4i v4i_lds;     // LDS-side pointee
#else
#define HAVE_ASYNC_LDS 0
#endif

__device__ __forceinline__ void wait_async0() {
#if __has_builtin(__builtin_amdgcn_s_wait_asynccnt)
  __builtin_amdgcn_s_wait_asynccnt(0);
#else
  asm volatile("s_wait_asynccnt 0x0" ::: "memory");
#endif
}

// ---------------------------------------------------------------------------
// Embedding gather (float4 vectorized): h[m, :] = embed[ids[m], :]
// ---------------------------------------------------------------------------
__global__ __launch_bounds__(256)
void embed_kernel(const int* __restrict__ ids, const float* __restrict__ emb,
                  float* __restrict__ h) {
  size_t idx4 = (size_t)blockIdx.x * 256 + threadIdx.x;  // over MTOK*DMODEL/4
  int m  = (int)(idx4 / (DMODEL / 4));
  int d4 = (int)(idx4 % (DMODEL / 4));
  reinterpret_cast<float4*>(h)[idx4] =
      reinterpret_cast<const float4*>(emb + (size_t)ids[m] * DMODEL)[d4];
}

// ---------------------------------------------------------------------------
// RMSNorm: one block per row, f16 or f32 output
// ---------------------------------------------------------------------------
template <bool F16OUT>
__global__ __launch_bounds__(256)
void rms_kernel(const float* __restrict__ x, const float* __restrict__ w,
                f16* __restrict__ o16, float* __restrict__ o32) {
  __shared__ float red[256];
  const int row = blockIdx.x;
  const float* xr = x + (size_t)row * DMODEL;
  float ss = 0.f;
  for (int d = threadIdx.x; d < DMODEL; d += 256) { float v = xr[d]; ss += v * v; }
  red[threadIdx.x] = ss;
  __syncthreads();
  for (int s = 128; s > 0; s >>= 1) {
    if ((int)threadIdx.x < s) red[threadIdx.x] += red[threadIdx.x + s];
    __syncthreads();
  }
  const float r = rsqrtf(red[0] / (float)DMODEL + RMS_EPS);
  for (int d = threadIdx.x; d < DMODEL; d += 256) {
    float v = xr[d] * r * w[d];
    if (F16OUT) o16[(size_t)row * DMODEL + d] = (f16)v;
    else        o32[(size_t)row * DMODEL + d] = v;
  }
}

// ---------------------------------------------------------------------------
// int8 -> f16 weight dequant, 16 elements per thread
// ---------------------------------------------------------------------------
__global__ __launch_bounds__(256)
void dequant_kernel(const int8_t* __restrict__ q, f16* __restrict__ o) {
  size_t i = ((size_t)blockIdx.x * 256 + threadIdx.x) * 16;
  int4 raw = *reinterpret_cast<const int4*>(q + i);
  const int8_t* bytes = reinterpret_cast<const int8_t*>(&raw);
  v16h out;
#pragma unroll
  for (int j = 0; j < 16; ++j) out[j] = (f16)(float)bytes[j];
  *reinterpret_cast<v16h*>(o + i) = out;
}

// ---------------------------------------------------------------------------
// WMMA GEMM: C[M,N] = scale[n] * sum_k A[m,k] * W[n,k]  (+ optional residual)
// Block = 8 waves as 4(M) x 2(N), block tile 256x128; each wave owns a 64x64
// C macro-tile (4x4 WMMA subtiles), K-step 32 of v_wmma_f32_16x16x32_f16.
//
// HAVE_ASYNC_LDS: A/B K-slabs are double-buffered in LDS via
// global_load_async_to_lds_b128 (ASYNCcnt), published with
// s_wait_asynccnt + barrier; fragments reload from LDS at WGP bandwidth.
// Fallback: direct-global fragment loads.
// ---------------------------------------------------------------------------
#define KS   32
#define LPAD 8   // pad LDS rows to 40 halves (80B) to spread banks

template <bool F16OUT, bool RESID>
__global__ __launch_bounds__(256)
void gemm_wmma_kernel(const f16* __restrict__ A, const f16* __restrict__ W,
                      const float* __restrict__ scale,
                      f16* __restrict__ out16, float* __restrict__ out32,
                      int Nn, int Kk) {
  const int lane  = threadIdx.x & 31;
  const int wave  = threadIdx.x >> 5;
  const int wm    = wave >> 1;                 // 0..3
  const int wn    = wave & 1;                  // 0..1
  const int g     = lane & 15;
  const int khalf = (lane >> 4) << 4;
  const int rowblk = blockIdx.y * 256;
  const int colblk = blockIdx.x * 128;

  v8f zero = {};
  v8f acc[4][4];
#pragma unroll
  for (int i = 0; i < 4; ++i)
#pragma unroll
    for (int j = 0; j < 4; ++j) acc[i][j] = zero;

#if HAVE_ASYNC_LDS
  __shared__ f16 As[2][256][KS + LPAD];
  __shared__ f16 Bs[2][128][KS + LPAD];
  const int t = threadIdx.x;
  // staging map: thread t owns A row t (64B = 4x b128); B row t&127, seg t>>7 (32B)
  const f16* ag = A + (size_t)(rowblk + t) * Kk;
  const int  brow = t & 127, bseg = (t >> 7) * 16;
  const f16* bg = W + (size_t)(colblk + brow) * Kk + bseg;

  auto stage = [&](int buf, int k0) {
    const f16* agk = ag + k0;
    f16* alb = &As[buf][t][0];
#pragma unroll
    for (int c = 0; c < 4; ++c)
      __builtin_amdgcn_global_load_async_to_lds_b128(
          (v4i*)(agk + c * 8), (v4i_lds*)(alb + c * 8), 0, 0);
    const f16* bgk = bg + k0;
    f16* blb = &Bs[buf][brow][bseg];
#pragma unroll
    for (int c = 0; c < 2; ++c)
      __builtin_amdgcn_global_load_async_to_lds_b128(
          (v4i*)(bgk + c * 8), (v4i_lds*)(blb + c * 8), 0, 0);
  };

  stage(0, 0);
  wait_async0();
  __syncthreads();

  const int nk = Kk / KS;
  for (int ki = 0; ki < nk; ++ki) {
    const int buf = ki & 1;
    if (ki + 1 < nk) stage(buf ^ 1, (ki + 1) * KS);

    v16h af[4], bf[4];
#pragma unroll
    for (int i = 0; i < 4; ++i)
      af[i] = *(const v16h*)(&As[buf][wm * 64 + i * 16 + g][khalf]);
#pragma unroll
    for (int j = 0; j < 4; ++j)
      bf[j] = *(const v16h*)(&Bs[buf][wn * 64 + j * 16 + g][khalf]);
#pragma unroll
    for (int i = 0; i < 4; ++i)
#pragma unroll
      for (int j = 0; j < 4; ++j)
        acc[i][j] = __builtin_amdgcn_wmma_f32_16x16x32_f16(
            false, af[i], false, bf[j], (short)0, acc[i][j], false, false);

    wait_async0();       // next buffer resident before anyone reads it
    __syncthreads();
  }
#else
  const f16* ap[4];
  const f16* wp[4];
#pragma unroll
  for (int i = 0; i < 4; ++i)
    ap[i] = A + (size_t)(rowblk + wm * 64 + i * 16 + g) * Kk + khalf;
#pragma unroll
  for (int j = 0; j < 4; ++j)
    wp[j] = W + (size_t)(colblk + wn * 64 + j * 16 + g) * Kk + khalf;

  for (int k = 0; k < Kk; k += 32) {
    v16h af[4], bf[4];
#pragma unroll
    for (int i = 0; i < 4; ++i) af[i] = *(const v16h*)(ap[i] + k);
#pragma unroll
    for (int j = 0; j < 4; ++j) bf[j] = *(const v16h*)(wp[j] + k);
    __builtin_prefetch(ap[0] + k + 512, 0, 0);   // global_prefetch_b8
    __builtin_prefetch(wp[0] + k + 512, 0, 0);
#pragma unroll
    for (int i = 0; i < 4; ++i)
#pragma unroll
      for (int j = 0; j < 4; ++j)
        acc[i][j] = __builtin_amdgcn_wmma_f32_16x16x32_f16(
            false, af[i], false, bf[j], (short)0, acc[i][j], false, false);
  }
#endif

  const int halfsel = (lane >> 4) << 3;
#pragma unroll
  for (int j = 0; j < 4; ++j) {
    const int col = colblk + wn * 64 + j * 16 + g;
    const float s = scale[col];
#pragma unroll
    for (int i = 0; i < 4; ++i) {
      const int row0 = rowblk + wm * 64 + i * 16 + halfsel;
#pragma unroll
      for (int r = 0; r < 8; ++r) {
        const size_t o = (size_t)(row0 + r) * Nn + col;
        float v = acc[i][j][r] * s;
        if (RESID) v += out32[o];
        if (F16OUT) out16[o] = (f16)v;
        else        out32[o] = v;
      }
    }
  }
}

// ---------------------------------------------------------------------------
// LoRA pass 1: t[m, r] = sum_d xn[m,d] * a[r,d]   (R = 8)
// ---------------------------------------------------------------------------
__global__ __launch_bounds__(256)
void lora1_kernel(const f16* __restrict__ xn, const float* __restrict__ a,
                  float* __restrict__ t) {
  __shared__ float red[256];
  const int row = blockIdx.x;
  float acc[LRANK];
#pragma unroll
  for (int r = 0; r < LRANK; ++r) acc[r] = 0.f;
  for (int d = threadIdx.x; d < DMODEL; d += 256) {
    float xv = (float)xn[(size_t)row * DMODEL + d];
#pragma unroll
    for (int r = 0; r < LRANK; ++r) acc[r] += xv * a[r * DMODEL + d];
  }
#pragma unroll
  for (int r = 0; r < LRANK; ++r) {
    red[threadIdx.x] = acc[r];
    __syncthreads();
    for (int s = 128; s > 0; s >>= 1) {
      if ((int)threadIdx.x < s) red[threadIdx.x] += red[threadIdx.x + s];
      __syncthreads();
    }
    if (threadIdx.x == 0) t[row * LRANK + r] = red[0];
    __syncthreads();
  }
}

// LoRA pass 2: o[m, c] += LORA_SCALE * sum_r t[m,r] * bm[c,r]  (f16 rmw)
__global__ __launch_bounds__(256)
void lora2_kernel(const float* __restrict__ t, const float* __restrict__ bm,
                  f16* __restrict__ o) {
  size_t idx = (size_t)blockIdx.x * 256 + threadIdx.x;   // over MTOK*DMODEL
  int m = (int)(idx / DMODEL);
  int c = (int)(idx % DMODEL);
  float v = 0.f;
#pragma unroll
  for (int r = 0; r < LRANK; ++r) v += t[m * LRANK + r] * bm[c * LRANK + r];
  o[idx] = (f16)((float)o[idx] + v * LORA_SCALE);
}

// ---------------------------------------------------------------------------
// RoPE (llama rotate_half), in place on f16 [B,S,H,HD]
// ---------------------------------------------------------------------------
__global__ __launch_bounds__(256)
void rope_kernel(f16* __restrict__ x) {
  size_t idx = (size_t)blockIdx.x * 256 + threadIdx.x;   // over MTOK*NHEAD*64
  int i  = (int)(idx & 63);
  int hh = (int)((idx >> 6) & (NHEAD - 1));
  int m  = (int)(idx >> 10);
  int s  = m & (SEQ - 1);
  float freq = __expf(-(float)(2 * i) * (LOG_THETA / (float)HEADD));
  float ang = (float)s * freq;
  float sn, cs;
  __sincosf(ang, &sn, &cs);
  size_t basei = (size_t)m * DMODEL + hh * HEADD + i;
  float x1 = (float)x[basei], x2 = (float)x[basei + 64];
  x[basei]      = (f16)(x1 * cs - x2 * sn);
  x[basei + 64] = (f16)(x2 * cs + x1 * sn);
}

// ---------------------------------------------------------------------------
// V transpose: vh [B,S,H,HD] -> vt [B,H,HD,S] so P*V B-fragments are contiguous
// ---------------------------------------------------------------------------
__global__ __launch_bounds__(256)
void vtrans_kernel(const f16* __restrict__ v, f16* __restrict__ vt) {
  size_t idx = (size_t)blockIdx.x * 256 + threadIdx.x;   // over MTOK*DMODEL
  int m = (int)(idx / DMODEL);
  int d = (int)(idx % DMODEL);
  int b = m >> 10, s = m & (SEQ - 1);
  int hh = d >> 7, dd = d & 127;
  vt[(((size_t)(b * NHEAD + hh) * HEADD) + dd) * SEQ + s] = v[idx];
}

// ---------------------------------------------------------------------------
// Flash attention: one wave per (b, h, 16-row q tile). QK^T and P*V on WMMA.
// Online softmax; P relayout C->A via LDS bounce.
// ---------------------------------------------------------------------------
__global__ __launch_bounds__(32)
void flash_attn_kernel(const f16* __restrict__ q, const f16* __restrict__ k,
                       const f16* __restrict__ vt, const int* __restrict__ mask,
                       f16* __restrict__ o) {
  __shared__ f16 plds[16 * 32];
  const int lane = threadIdx.x;
  const int g    = lane & 15;
  const int half = lane >> 4;
  const int QT   = SEQ / 16;
  const int qt   = blockIdx.x % QT;
  const int bh   = blockIdx.x / QT;
  const int hh   = bh % NHEAD;
  const int b    = bh / NHEAD;
  const int qbase = qt * 16;

  v16h qf[4];
  const f16* qrow = q + (size_t)(b * SEQ + qbase + g) * DMODEL + hh * HEADD;
#pragma unroll
  for (int c = 0; c < 4; ++c) qf[c] = *(const v16h*)(qrow + c * 32 + half * 16);

  v8f zero = {};
  v8f ctx[8];
#pragma unroll
  for (int n = 0; n < 8; ++n) ctx[n] = zero;
  float mst[8], lst[8];
#pragma unroll
  for (int r = 0; r < 8; ++r) { mst[r] = -1e30f; lst[r] = 0.f; }

  const int kend = qbase + 16;                  // causal upper bound (exclusive)
  for (int kt = 0; kt < kend; kt += 32) {
    // ---- scores for two adjacent 16-col key tiles ----
    v8f s0 = zero, s1 = zero;
    const f16* k0p = k + (size_t)(b * SEQ + kt + g) * DMODEL + hh * HEADD;
    const f16* k1p = k0p + (size_t)16 * DMODEL;
#pragma unroll
    for (int c = 0; c < 4; ++c) {
      v16h k0 = *(const v16h*)(k0p + c * 32 + half * 16);
      v16h k1 = *(const v16h*)(k1p + c * 32 + half * 16);
      s0 = __builtin_amdgcn_wmma_f32_16x16x32_f16(false, qf[c], false, k0,
                                                  (short)0, s0, false, false);
      s1 = __builtin_amdgcn_wmma_f32_16x16x32_f16(false, qf[c], false, k1,
                                                  (short)0, s1, false, false);
    }
    const int col0 = kt + g;
    const int col1 = kt + 16 + g;
    const float mv0 = (mask[b * SEQ + col0] > 0) ? 0.f : -1e9f;
    const float mv1 = (mask[b * SEQ + col1] > 0) ? 0.f : -1e9f;

    // ---- online softmax (rows live across 16-lane groups) ----
    float p0[8], p1[8];
#pragma unroll
    for (int r = 0; r < 8; ++r) {
      const int row = qbase + r + half * 8;
      float x0 = s0[r] * INV_SQRT_HD + ((col0 <= row) ? mv0 : -1e9f);
      float x1 = s1[r] * INV_SQRT_HD + ((col1 <= row) ? mv1 : -1e9f);
      float mx = fmaxf(x0, x1);
#pragma unroll
      for (int off = 8; off > 0; off >>= 1) mx = fmaxf(mx, __shfl_xor(mx, off, 16));
      const float mnew = fmaxf(mst[r], mx);
      const float corr = __expf(mst[r] - mnew);
      const float e0 = __expf(x0 - mnew);
      const float e1 = __expf(x1 - mnew);
      float sm = e0 + e1;
#pragma unroll
      for (int off = 8; off > 0; off >>= 1) sm += __shfl_xor(sm, off, 16);
      lst[r] = lst[r] * corr + sm;
      mst[r] = mnew;
      p0[r] = e0;
      p1[r] = e1;
#pragma unroll
      for (int n = 0; n < 8; ++n) ctx[n][r] *= corr;
    }

    // ---- P relayout C->A via LDS (16x32 f16, row-major) ----
    __syncthreads();
#pragma unroll
    for (int r = 0; r < 8; ++r) {
      const int lrow = r + half * 8;
      plds[lrow * 32 + g]      = (f16)p0[r];
      plds[lrow * 32 + 16 + g] = (f16)p1[r];
    }
    __syncthreads();
    v16h pa = *(const v16h*)(&plds[g * 32 + half * 16]);

    // ---- ctx += P (16x32) * V (32x16 per HD chunk) ----
    const f16* vbase = vt + (size_t)(b * NHEAD + hh) * HEADD * SEQ;
#pragma unroll
    for (int n = 0; n < 8; ++n) {
      v16h vf = *(const v16h*)(vbase + (size_t)(n * 16 + g) * SEQ + kt + half * 16);
      ctx[n] = __builtin_amdgcn_wmma_f32_16x16x32_f16(false, pa, false, vf,
                                                      (short)0, ctx[n], false, false);
    }
  }

#pragma unroll
  for (int r = 0; r < 8; ++r) {
    const float inv = 1.0f / lst[r];
    const int row = qbase + r + half * 8;
    f16* orow = o + (size_t)(b * SEQ + row) * DMODEL + hh * HEADD;
#pragma unroll
    for (int n = 0; n < 8; ++n) orow[n * 16 + g] = (f16)(ctx[n][r] * inv);
  }
}

// ---------------------------------------------------------------------------
// SwiGLU activation: g = silu(g) * u  (in place on g, f16)
// ---------------------------------------------------------------------------
__global__ __launch_bounds__(256)
void silumul_kernel(f16* __restrict__ gbuf, const f16* __restrict__ ubuf) {
  size_t idx = (size_t)blockIdx.x * 256 + threadIdx.x;   // over MTOK*FFN
  float gv = (float)gbuf[idx];
  float uv = (float)ubuf[idx];
  gbuf[idx] = (f16)(gv / (1.f + __expf(-gv)) * uv);
}

// ---------------------------------------------------------------------------
// Masked mean pool: out[b, d] = sum_s mask * hf[b,s,d] / max(sum mask, 1)
// ---------------------------------------------------------------------------
__global__ __launch_bounds__(256)
void pool_kernel(const float* __restrict__ hf, const int* __restrict__ mask,
                 float* __restrict__ out) {
  int idx = blockIdx.x * 256 + threadIdx.x;   // over BATCH*DMODEL
  int b = idx / DMODEL;
  int d = idx % DMODEL;
  float sum = 0.f, cnt = 0.f;
  for (int s = 0; s < SEQ; ++s) {
    if (mask[b * SEQ + s] > 0) {
      sum += hf[(size_t)(b * SEQ + s) * DMODEL + d];
      cnt += 1.f;
    }
  }
  out[idx] = sum / fmaxf(cnt, 1.f);
}

// ---------------------------------------------------------------------------
// Host orchestration
// ---------------------------------------------------------------------------
extern "C" void kernel_launch(void* const* d_in, const int* in_sizes, int n_in,
                              void* d_out, int out_size, void* d_ws, size_t ws_size,
                              hipStream_t stream) {
  (void)in_sizes; (void)n_in; (void)out_size; (void)ws_size;

  const int*    ids   = (const int*)d_in[0];
  const int*    amask = (const int*)d_in[1];
  const float*  emb   = (const float*)d_in[2];
  const int8_t* wq_q  = (const int8_t*)d_in[3];
  const float*  wq_s  = (const float*)d_in[4];
  const int8_t* wk_q  = (const int8_t*)d_in[5];
  const float*  wk_s  = (const float*)d_in[6];
  const int8_t* wv_q  = (const int8_t*)d_in[7];
  const float*  wv_s  = (const float*)d_in[8];
  const int8_t* wo_q  = (const int8_t*)d_in[9];
  const float*  wo_s  = (const float*)d_in[10];
  const int8_t* wg_q  = (const int8_t*)d_in[11];
  const float*  wg_s  = (const float*)d_in[12];
  const int8_t* wu_q  = (const int8_t*)d_in[13];
  const float*  wu_s  = (const float*)d_in[14];
  const int8_t* wd_q  = (const int8_t*)d_in[15];
  const float*  wd_s  = (const float*)d_in[16];
  const float*  la_q  = (const float*)d_in[17];
  const float*  lb_q  = (const float*)d_in[18];
  const float*  la_v  = (const float*)d_in[19];
  const float*  lb_v  = (const float*)d_in[20];
  const float*  ln1   = (const float*)d_in[21];
  const float*  ln2   = (const float*)d_in[22];
  const float*  lnf   = (const float*)d_in[23];
  float* outp = (float*)d_out;

  // ---- workspace carve (aligned 256B) ----
  char* base = (char*)d_ws;
  size_t off = 0;
  auto take = [&](size_t bytes) -> char* {
    char* p = base + off;
    off += (bytes + 255) & ~(size_t)255;
    return p;
  };
  float* h     = (float*)take((size_t)MTOK * DMODEL * 4);
  f16*   xn16  = (f16*)  take((size_t)MTOK * DMODEL * 2);
  f16*   qh    = (f16*)  take((size_t)MTOK * DMODEL * 2);
  f16*   kh    = (f16*)  take((size_t)MTOK * DMODEL * 2);
  f16*   vh    = (f16*)  take((size_t)MTOK * DMODEL * 2);
  f16*   vtb   = (f16*)  take((size_t)MTOK * DMODEL * 2);
  f16*   cx    = (f16*)  take((size_t)MTOK * DMODEL * 2);
  f16*   g16   = (f16*)  take((size_t)MTOK * FFN * 2);
  f16*   u16   = (f16*)  take((size_t)MTOK * FFN * 2);
  float* loraT = (float*)take((size_t)MTOK * LRANK * 4);
  f16*   wbuf  = (f16*)  take((size_t)FFN * DMODEL * 2);
  float* hf    = (float*)u16;   // final-norm f32 aliases dead 'up' buffer

  const dim3 B256(256), B32(32);
  const int GB_MD    = (MTOK * DMODEL) / 256;          // 32768
  const int GB_MD4   = (MTOK * DMODEL) / (256 * 4);    // 8192 (float4 embed)
  const int GB_DD16  = (DMODEL * DMODEL) / (256 * 16); // 1024 (16B dequant)
  const int GB_FD16  = (FFN * DMODEL) / (256 * 16);    // 2752
  const int GB_MF    = (MTOK * FFN) / 256;             // 88064
  const int GB_ROPE  = (MTOK * NHEAD * 64) / 256;      // 16384
  const int GB_ATTN  = BATCH * NHEAD * (SEQ / 16);     // 4096
  const dim3 GEMM_D(DMODEL / 128, MTOK / 256);         // 16 x 16
  const dim3 GEMM_F(FFN / 128,    MTOK / 256);         // 43 x 16

  embed_kernel<<<GB_MD4, B256, 0, stream>>>(ids, emb, h);

  for (int l = 0; l < LAYERS; ++l) {
    const size_t oDD = (size_t)l * DMODEL * DMODEL;
    const size_t oFD = (size_t)l * FFN * DMODEL;

    rms_kernel<true><<<MTOK, B256, 0, stream>>>(h, ln1 + l * DMODEL, xn16, nullptr);

    // Q = deq_mm + LoRA
    dequant_kernel<<<GB_DD16, B256, 0, stream>>>(wq_q + oDD, wbuf);
    gemm_wmma_kernel<true, false><<<GEMM_D, B256, 0, stream>>>(
        xn16, wbuf, wq_s + l * DMODEL, qh, nullptr, DMODEL, DMODEL);
    lora1_kernel<<<MTOK, B256, 0, stream>>>(xn16, la_q + (size_t)l * LRANK * DMODEL, loraT);
    lora2_kernel<<<GB_MD, B256, 0, stream>>>(loraT, lb_q + (size_t)l * DMODEL * LRANK, qh);

    // K
    dequant_kernel<<<GB_DD16, B256, 0, stream>>>(wk_q + oDD, wbuf);
    gemm_wmma_kernel<true, false><<<GEMM_D, B256, 0, stream>>>(
        xn16, wbuf, wk_s + l * DMODEL, kh, nullptr, DMODEL, DMODEL);

    // V = deq_mm + LoRA
    dequant_kernel<<<GB_DD16, B256, 0, stream>>>(wv_q + oDD, wbuf);
    gemm_wmma_kernel<true, false><<<GEMM_D, B256, 0, stream>>>(
        xn16, wbuf, wv_s + l * DMODEL, vh, nullptr, DMODEL, DMODEL);
    lora1_kernel<<<MTOK, B256, 0, stream>>>(xn16, la_v + (size_t)l * LRANK * DMODEL, loraT);
    lora2_kernel<<<GB_MD, B256, 0, stream>>>(loraT, lb_v + (size_t)l * DMODEL * LRANK, vh);

    // RoPE + attention
    rope_kernel<<<GB_ROPE, B256, 0, stream>>>(qh);
    rope_kernel<<<GB_ROPE, B256, 0, stream>>>(kh);
    vtrans_kernel<<<GB_MD, B256, 0, stream>>>(vh, vtb);
    flash_attn_kernel<<<GB_ATTN, B32, 0, stream>>>(qh, kh, vtb, amask, cx);

    // O-proj with fused residual into h
    dequant_kernel<<<GB_DD16, B256, 0, stream>>>(wo_q + oDD, wbuf);
    gemm_wmma_kernel<false, true><<<GEMM_D, B256, 0, stream>>>(
        cx, wbuf, wo_s + l * DMODEL, nullptr, h, DMODEL, DMODEL);

    // MLP
    rms_kernel<true><<<MTOK, B256, 0, stream>>>(h, ln2 + l * DMODEL, xn16, nullptr);
    dequant_kernel<<<GB_FD16, B256, 0, stream>>>(wg_q + oFD, wbuf);
    gemm_wmma_kernel<true, false><<<GEMM_F, B256, 0, stream>>>(
        xn16, wbuf, wg_s + l * FFN, g16, nullptr, FFN, DMODEL);
    dequant_kernel<<<GB_FD16, B256, 0, stream>>>(wu_q + oFD, wbuf);
    gemm_wmma_kernel<true, false><<<GEMM_F, B256, 0, stream>>>(
        xn16, wbuf, wu_s + l * FFN, u16, nullptr, FFN, DMODEL);
    silumul_kernel<<<GB_MF, B256, 0, stream>>>(g16, u16);
    dequant_kernel<<<GB_FD16, B256, 0, stream>>>(wd_q + oFD, wbuf);   // wd is [D, F]
    gemm_wmma_kernel<false, true><<<GEMM_D, B256, 0, stream>>>(
        g16, wbuf, wd_s + l * DMODEL, nullptr, h, DMODEL, FFN);
  }

  rms_kernel<false><<<MTOK, B256, 0, stream>>>(h, lnf, nullptr, hf);
  pool_kernel<<<(BATCH * DMODEL) / 256, B256, 0, stream>>>(hf, amask, outp);
}
